// GAT_23931557773316
// MI455X (gfx1250) — matrix operand order repaired and checked by reading, not compiled, not run on hardware
//
#include <hip/hip_runtime.h>
#include <math.h>

// Problem constants (from reference)
#define NN    50000
#define EE    800000
#define HID   64
#define HEADS 4
#define NCLS  40
#define SLOPE 0.2f

typedef __attribute__((ext_vector_type(16))) _Float16 v16h;
typedef __attribute__((ext_vector_type(8)))  _Float16 v8h;
typedef __attribute__((ext_vector_type(4)))  _Float16 v4h;
typedef __attribute__((ext_vector_type(8)))  float    v8f;

// ---------------------------------------------------------------------------
// WMMA GEMM: C[M,Nout] = A[M,256] @ B[256,Nout], f32 in/out, f16 mul, f32 acc
// Block: 256 threads = 8 waves; block tile 64(M) x 32(N); wave tile 16x16.
// Software-pipelined: next tile's global loads issue before the WMMA.
// ---------------------------------------------------------------------------
#define TM 64
#define TN 32
#define TK 32
#define LDSPAD 40   // halfs per LDS row (80 B: keeps 16 B alignment, skews banks)

__global__ __launch_bounds__(256)
void gemm_wmma(const float* __restrict__ A, const float* __restrict__ B,
               float* __restrict__ C, int M, int Nout) {
    constexpr int K = 256;
    __shared__ _Float16 As[TM][LDSPAD];
    __shared__ _Float16 Bs[TN][LDSPAD];   // stored transposed: Bs[n][k]

    const int tid  = threadIdx.x;
    const int wave = tid >> 5;            // wave32
    const int lane = tid & 31;
    const int blockM = blockIdx.x * TM;
    const int blockN = blockIdx.y * TN;

    const int mTile = wave >> 1;          // 0..3
    const int nTile = wave & 1;           // 0..1
    const int halfSel = lane >> 4;        // 0 or 1

    // A staging role: row = tid/4, 8-float segment = tid%4 (2x float4, b128)
    const int arow = tid >> 2, aseg = tid & 3;
    int gr = blockM + arow;
    if (gr >= M) gr = M - 1;              // clamp: no EXEC divergence in loop
    const float* aBase = A + (size_t)gr * K + aseg * 8;

    // B staging role: n = tid%32, 4 k-rows = wave*4.. (each k-row coalesced)
    const int bn = tid & 31, bk = wave * 4;
    const float* bBase = B + (size_t)bk * Nout + blockN + bn;

    float4 ra0, ra1;
    float  rb0, rb1, rb2, rb3;
    auto loadTile = [&](int k0) {
        const float4* ap = (const float4*)(aBase + k0);
        ra0 = ap[0];
        ra1 = ap[1];
        const float* bp = bBase + (size_t)k0 * Nout;
        rb0 = bp[0];
        rb1 = bp[(size_t)Nout];
        rb2 = bp[(size_t)2 * Nout];
        rb3 = bp[(size_t)3 * Nout];
    };

    loadTile(0);
    v8f acc = {};

    for (int k0 = 0; k0 < K; k0 += TK) {
        __syncthreads();   // LDS free (prev fragments consumed)
        // Stage registers -> LDS as f16 (one b128 + one b64 store per thread)
        v8h ah;
        ah[0] = (_Float16)ra0.x; ah[1] = (_Float16)ra0.y;
        ah[2] = (_Float16)ra0.z; ah[3] = (_Float16)ra0.w;
        ah[4] = (_Float16)ra1.x; ah[5] = (_Float16)ra1.y;
        ah[6] = (_Float16)ra1.z; ah[7] = (_Float16)ra1.w;
        *(v8h*)&As[arow][aseg * 8] = ah;
        v4h bh;
        bh[0] = (_Float16)rb0; bh[1] = (_Float16)rb1;
        bh[2] = (_Float16)rb2; bh[3] = (_Float16)rb3;
        *(v4h*)&Bs[bn][bk] = bh;
        __syncthreads();

        // Prefetch next tile while this tile is consumed (overlaps WMMA)
        if (k0 + TK < K) loadTile(k0 + TK);

        // A fragment (16-bit A 16x32 layout):
        //   lane<16: elems 0..7 = K 0..7,  8..15 = K 16..23
        //   lane>=16: elems 0..7 = K 8..15, 8..15 = K 24..31
        const int mrow = (mTile << 4) + (lane & 15);
        const int kA = halfSel * 8;
        v8h a0 = *(const v8h*)&As[mrow][kA];
        v8h a1 = *(const v8h*)&As[mrow][kA + 16];
        v16h aFrag = __builtin_shufflevector(a0, a1,
            0, 1, 2, 3, 4, 5, 6, 7, 8, 9, 10, 11, 12, 13, 14, 15);

        // B fragment (32x16): lane holds col N=lane%16, same K order as A:
        //   lane<16: elems 0..7 = K 0..7, 8..15 = K 16..23; lane>=16: K 8..15, 24..31.
        const int ncol = (nTile << 4) + (lane & 15);
        const int kB = halfSel * 8;
        v8h b0 = *(const v8h*)&Bs[ncol][kB];
        v8h b1 = *(const v8h*)&Bs[ncol][kB + 16];
        v16h bFrag = __builtin_shufflevector(b0, b1,
            0, 1, 2, 3, 4, 5, 6, 7, 8, 9, 10, 11, 12, 13, 14, 15);

        acc = __builtin_amdgcn_wmma_f32_16x16x32_f16(
            /*neg_a=*/false, aFrag, /*neg_b=*/false, bFrag,
            /*c_mod=*/(short)0, acc, /*reuse_a=*/false, /*reuse_b=*/false);
    }

    // C/D layout: VGPR r -> M = r + 8*(lane/16), N = lane%16.
    const int mBase = blockM + (mTile << 4) + (halfSel << 3);
    const int n = blockN + (nTile << 4) + (lane & 15);
    #pragma unroll
    for (int r = 0; r < 8; ++r) {
        int gm = mBase + r;
        if (gm < M) C[(size_t)gm * Nout + n] = acc[r];
    }
}

// ---------------------------------------------------------------------------
// Attention scores: si[n,h] = <h[n,h,:], ai[h,:]>, sj likewise.
// ---------------------------------------------------------------------------
__global__ void att_scores(const float* __restrict__ h,
                           const float* __restrict__ ai,
                           const float* __restrict__ aj,
                           float* __restrict__ si, float* __restrict__ sj, int F) {
    int idx = blockIdx.x * blockDim.x + threadIdx.x;   // n*HEADS + head
    if (idx >= NN * HEADS) return;
    int n = idx >> 2, hd = idx & 3;
    const float* row = h + (size_t)n * HEADS * F + (size_t)hd * F;
    const float* av = ai + hd * F;
    const float* bv = aj + hd * F;
    float a = 0.f, b = 0.f;
    for (int f = 0; f < F; ++f) { float v = row[f]; a += v * av[f]; b += v * bv[f]; }
    si[idx] = a; sj[idx] = b;
}

__global__ void fill_f32(float* __restrict__ p, float v, int n) {
    int i = blockIdx.x * blockDim.x + threadIdx.x;
    if (i < n) p[i] = v;
}

// Float atomic max via sign-magnitude int ordering (buffer init to -inf).
__device__ inline void atomicMaxF32(float* addr, float val) {
    if (val >= 0.0f) atomicMax((int*)addr, __float_as_int(val));
    else             atomicMin((unsigned int*)addr, __float_as_uint(val));
}

// Pass 1: e = leaky_relu(si[dst] + sj[src]); segment max over dst.
__global__ void edge_lrelu_max(const int* __restrict__ src, const int* __restrict__ dst,
                               const float* __restrict__ si, const float* __restrict__ sj,
                               float* __restrict__ mx, float* __restrict__ eb) {
    int idx = blockIdx.x * blockDim.x + threadIdx.x;   // e*HEADS + head
    if (idx >= EE * HEADS) return;
    int e = idx >> 2, hd = idx & 3;
    int s = src[e], d = dst[e];
    float v = si[d * HEADS + hd] + sj[s * HEADS + hd];
    v = (v > 0.0f) ? v : SLOPE * v;
    eb[idx] = v;
    atomicMaxF32(&mx[d * HEADS + hd], v);
}

// Pass 2: ex = exp(e - m[dst]); segment sum into denom.
__global__ void edge_exp_sum(const int* __restrict__ dst,
                             const float* __restrict__ mx, float* __restrict__ dn,
                             float* __restrict__ eb) {
    int idx = blockIdx.x * blockDim.x + threadIdx.x;
    if (idx >= EE * HEADS) return;
    int e = idx >> 2, hd = idx & 3;
    int d = dst[e];
    float ex = expf(eb[idx] - mx[d * HEADS + hd]);
    eb[idx] = ex;
    atomicAdd(&dn[d * HEADS + hd], ex);
}

// Pass 3: out[dst] += h[src] * (ex/denom[dst]); 4 channels per thread.
__global__ void edge_aggr(const int* __restrict__ src, const int* __restrict__ dst,
                          const float* __restrict__ h, const float* __restrict__ eb,
                          const float* __restrict__ dn, float* __restrict__ out, int F) {
    const int D = HEADS * F;
    const int DQ = D >> 2;                 // channel quads per edge
    long long idx = (long long)blockIdx.x * blockDim.x + threadIdx.x;
    if (idx >= (long long)EE * DQ) return;
    int e  = (int)(idx / DQ);
    int cq = (int)(idx - (long long)e * DQ);
    int c  = cq << 2;
    int hd = c / F;                        // F multiple of 4: quad within one head
    int s = src[e], d = dst[e];
    float alpha = eb[e * HEADS + hd] / dn[d * HEADS + hd];
    float4 hv = *(const float4*)(h + (size_t)s * D + c);
    float* op = out + (size_t)d * D + c;
    atomicAdd(op + 0, hv.x * alpha);
    atomicAdd(op + 1, hv.y * alpha);
    atomicAdd(op + 2, hv.z * alpha);
    atomicAdd(op + 3, hv.w * alpha);
}

__global__ void elu_inplace(float* __restrict__ p, long long n) {
    long long i = (long long)blockIdx.x * blockDim.x + threadIdx.x;
    if (i >= n) return;
    float v = p[i];
    p[i] = (v > 0.0f) ? v : expm1f(v);
}

// Final: mean over heads then log_softmax over NCLS.
__global__ void mean_logsoftmax(const float* __restrict__ h, float* __restrict__ out) {
    int n = blockIdx.x * blockDim.x + threadIdx.x;
    if (n >= NN) return;
    const float* r = h + (size_t)n * HEADS * NCLS;
    float v[NCLS];
    float mx = -3.4e38f;
    #pragma unroll
    for (int c = 0; c < NCLS; ++c) {
        float s = 0.f;
        #pragma unroll
        for (int hd = 0; hd < HEADS; ++hd) s += r[hd * NCLS + c];
        s *= (1.0f / HEADS);
        v[c] = s;
        mx = fmaxf(mx, s);
    }
    float se = 0.f;
    #pragma unroll
    for (int c = 0; c < NCLS; ++c) se += expf(v[c] - mx);
    float lse = mx + logf(se);
    #pragma unroll
    for (int c = 0; c < NCLS; ++c) out[(size_t)n * NCLS + c] = v[c] - lse;
}

// ---------------------------------------------------------------------------
// One GAT layer: gemm -> scores -> softmax(3 edge passes) -> aggregate.
// ---------------------------------------------------------------------------
static void gat_layer(const float* xin, const float* W, const float* ai, const float* aj,
                      const int* src, const int* dst,
                      float* H, float* O, float* SI, float* SJ,
                      float* MX, float* DN, float* EB,
                      int F, hipStream_t stream) {
    const int D = HEADS * F;
    dim3 gg((NN + TM - 1) / TM, D / TN);
    gemm_wmma<<<gg, 256, 0, stream>>>(xin, W, H, NN, D);

    const int nh = NN * HEADS;
    att_scores<<<(nh + 255) / 256, 256, 0, stream>>>(H, ai, aj, SI, SJ, F);
    fill_f32<<<(nh + 255) / 256, 256, 0, stream>>>(MX, -__builtin_inff(), nh);
    hipMemsetAsync(DN, 0, sizeof(float) * (size_t)nh, stream);
    hipMemsetAsync(O, 0, sizeof(float) * (size_t)NN * D, stream);

    const int eh = EE * HEADS;
    edge_lrelu_max<<<(eh + 255) / 256, 256, 0, stream>>>(src, dst, SI, SJ, MX, EB);
    edge_exp_sum<<<(eh + 255) / 256, 256, 0, stream>>>(dst, MX, DN, EB);
    long long tot = (long long)EE * (D >> 2);
    edge_aggr<<<(unsigned)((tot + 255) / 256), 256, 0, stream>>>(src, dst, H, EB, DN, O, F);
}

extern "C" void kernel_launch(void* const* d_in, const int* in_sizes, int n_in,
                              void* d_out, int out_size, void* d_ws, size_t ws_size,
                              hipStream_t stream) {
    const float* x   = (const float*)d_in[0];
    const int*   ei  = (const int*)d_in[1];
    const float* W1  = (const float*)d_in[2];
    const float* ai1 = (const float*)d_in[3];
    const float* aj1 = (const float*)d_in[4];
    const float* W2  = (const float*)d_in[5];
    const float* ai2 = (const float*)d_in[6];
    const float* aj2 = (const float*)d_in[7];
    const float* W3  = (const float*)d_in[8];
    const float* ai3 = (const float*)d_in[9];
    const float* aj3 = (const float*)d_in[10];
    const int* src = ei;        // edge_index[0] (source j)
    const int* dst = ei + EE;   // edge_index[1] (target i / segment id)
    float* out = (float*)d_out;

    // Workspace layout (floats)
    float* H  = (float*)d_ws;                  // [N,256] gemm output
    float* O  = H  + (size_t)NN * 256;         // [N,256] aggregated output
    float* SI = O  + (size_t)NN * 256;         // [N,4]
    float* SJ = SI + (size_t)NN * HEADS;       // [N,4]
    float* MX = SJ + (size_t)NN * HEADS;       // [N,4] segment max
    float* DN = MX + (size_t)NN * HEADS;       // [N,4] segment denom
    float* EB = DN + (size_t)NN * HEADS;       // [E,4] edge scores / exp

    const long long nd1 = (long long)NN * (HEADS * HID);   // N*256

    // Layer 1: x -> O, ELU
    gat_layer(x, W1, ai1, aj1, src, dst, H, O, SI, SJ, MX, DN, EB, HID, stream);
    elu_inplace<<<(unsigned)((nd1 + 255) / 256), 256, 0, stream>>>(O, nd1);

    // Layer 2: O -> O (gemm fully consumes O into H before O is re-zeroed), ELU
    gat_layer(O, W2, ai2, aj2, src, dst, H, O, SI, SJ, MX, DN, EB, HID, stream);
    elu_inplace<<<(unsigned)((nd1 + 255) / 256), 256, 0, stream>>>(O, nd1);

    // Layer 3: O -> O as [N, HEADS*NCLS]
    gat_layer(O, W3, ai3, aj3, src, dst, H, O, SI, SJ, MX, DN, EB, NCLS, stream);

    // mean over heads + log_softmax -> d_out [N, NCLS]
    mean_logsoftmax<<<(NN + 127) / 128, 128, 0, stream>>>(O, out);
}
